// HSTUAttention_79757542687326
// MI455X (gfx1250) — matrix operand order repaired and checked by reading, not compiled, not run on hardware
//
#include <hip/hip_runtime.h>
#include <hip/hip_bf16.h>

typedef __attribute__((ext_vector_type(16))) __bf16 v16bf;
typedef __attribute__((ext_vector_type(8)))  __bf16 v8bf;
typedef __attribute__((ext_vector_type(8)))  float  v8f;

#define SHUF16(lo, hi) __builtin_shufflevector((lo), (hi), 0,1,2,3,4,5,6,7,8,9,10,11,12,13,14,15)

// Problem constants
#define BB 4
#define SS 2048
#define DD 512
#define HH 8
#define HD 64
#define BS (BB*SS)          // 8192

// silu via v_exp_f32 + v_rcp_f32 (avoids the IEEE div-scale/div-fixup chain)
__device__ __forceinline__ float silu_f(float x) {
  return x * __builtin_amdgcn_rcpf(1.0f + __expf(-x));
}

// ---- Fragment loaders (CDNA5 wave32 WMMA VGPR layouts) ------------------
// A: 16x32 bf16, row-major source [M,K]. Lane L: row = L%16,
//    K-chunks at kb + (L/16)*8 + {0..7} and +{16..23}.
__device__ __forceinline__ v16bf load_a_frag(const __bf16* base, int lda,
                                             int row0, int kb, int lane) {
  int lm = lane & 15, lh = lane >> 4;
  const __bf16* p = base + (size_t)(row0 + lm) * lda + kb + lh * 8;
  v8bf lo = *(const v8bf*)p;
  v8bf hi = *(const v8bf*)(p + 16);
  return SHUF16(lo, hi);
}

// ---- Wave-level 64x64 GEMM mainloop: C += A(row-major) x B(N-major)^T ----
// K is a compile-time constant so per-tile row offsets are immediates and
// the only live address state is two incrementing base pointers.
template <int K>
__device__ __forceinline__ void wave_gemm_64x64(const __bf16* __restrict__ A,
                                                const __bf16* __restrict__ B,
                                                int row0, int col0,
                                                int lane, v8f acc[4][4]) {
  int lm = lane & 15, lh = lane >> 4;
  const __bf16* pa = A + (size_t)(row0 + lm) * K + lh * 8;
  const __bf16* pb = B + (size_t)(col0 + lm) * K + lh * 8;
#pragma unroll 1
  for (int kb = 0; kb < K; kb += 32) {
    v16bf af[4];
#pragma unroll
    for (int i = 0; i < 4; i++) {
      const __bf16* p = pa + (size_t)i * 16 * K;
      v8bf lo = *(const v8bf*)p;
      v8bf hi = *(const v8bf*)(p + 16);
      af[i] = SHUF16(lo, hi);
    }
#pragma unroll
    for (int j = 0; j < 4; j++) {
      const __bf16* p = pb + (size_t)j * 16 * K;
      v8bf lo = *(const v8bf*)p;
      v8bf hi = *(const v8bf*)(p + 16);
      v16bf bf = SHUF16(lo, hi);
#pragma unroll
      for (int i = 0; i < 4; i++)
        acc[i][j] = __builtin_amdgcn_wmma_f32_16x16x32_bf16(
            false, af[i], false, bf, (short)0, acc[i][j], false, false);
    }
    pa += 32;
    pb += 32;
  }
}

// ---- Kernel 0: f32 -> bf16 convert ---------------------------------------
__global__ void cvt_bf16_kernel(const float* __restrict__ in,
                                __bf16* __restrict__ out, int n) {
  int i = blockIdx.x * 256 + threadIdx.x;
  if (i < n) out[i] = (__bf16)in[i];
}

// ---- Kernel 1: h = silu(x@W1^T + b1), scatter to gate/q/k/vT -------------
// grid (64, 8), block 256 (8 waves, 2(M) x 4(N)), block tile 128x256.
__global__ __launch_bounds__(256, 1) void gemm1_kernel(
    const __bf16* __restrict__ xb, const __bf16* __restrict__ w1b,
    const float* __restrict__ b1, __bf16* __restrict__ qo,
    __bf16* __restrict__ ko, __bf16* __restrict__ vTo,
    float* __restrict__ gateo) {
  int wid = threadIdx.x >> 5, lane = threadIdx.x & 31;
  int row0 = blockIdx.x * 128 + (wid & 1) * 64;   // over B*S
  int col0 = blockIdx.y * 256 + (wid >> 1) * 64;  // over 4*D
  v8f acc[4][4] = {};
  wave_gemm_64x64<DD>(xb, w1b, row0, col0, lane, acc);

  int lm = lane & 15, lh = lane >> 4;
  int f = col0 >> 9;            // gate/q/k/v selector (wave-uniform)
  int hh = (col0 >> 6) & 7;     // head (wave-uniform)
#pragma unroll
  for (int j = 0; j < 4; j++) {
    int d = j * 16 + lm;        // 0..63 within head
    int n = col0 + j * 16 + lm;
    float bias = b1[n];
#pragma unroll
    for (int i = 0; i < 4; i++) {
#pragma unroll
      for (int e = 0; e < 8; e++) {
        int bs = row0 + i * 16 + e + lh * 8;
        float val = silu_f(acc[i][j][e] + bias);
        int b = bs >> 11, s = bs & (SS - 1);
        if (f == 0) {
          gateo[(size_t)bs * DD + hh * HD + d] = val;
        } else {
          __bf16 bv = (__bf16)val;
          size_t bhsd = (((size_t)(b * HH + hh) * SS) + s) * HD + d;
          if (f == 1)      qo[bhsd] = bv;
          else if (f == 2) ko[bhsd] = bv;
          else             vTo[(((size_t)(b * HH + hh) * HD) + d) * SS + s] = bv;
        }
      }
    }
  }
}

// ---- Kernel 2: causal SiLU attention, o = silu(q k^T * scale) @ v --------
// One wave per (b,h,16-row q block). grid 512, block 256.
__global__ __launch_bounds__(256, 1) void attn_kernel(
    const __bf16* __restrict__ q, const __bf16* __restrict__ k,
    const __bf16* __restrict__ vT, float* __restrict__ o) {
  __shared__ __bf16 smem[8][16 * 32];   // 1 KB per wave
  int wid = threadIdx.x >> 5, lane = threadIdx.x & 31;
  int task = blockIdx.x * 8 + wid;      // 0..4095
  int bh = task >> 7;                   // (b*H + h)
  int qb = task & 127;
  int b = bh >> 3, h = bh & 7;
  const __bf16* qp = q  + (size_t)bh * SS * HD;
  const __bf16* kp = k  + (size_t)bh * SS * HD;
  const __bf16* vp = vT + (size_t)bh * HD * SS;
  int s0 = qb * 16;
  int lm = lane & 15, lh = lane >> 4;
  __bf16* at = smem[wid];

  // q fragments for the two K=32 steps of hd=64 (resident whole loop)
  v16bf qfrag[2];
#pragma unroll
  for (int ks = 0; ks < 2; ks++) qfrag[ks] = load_a_frag(qp, HD, s0, ks * 32, lane);

  // incrementing base pointers for k (B-frag over t) and vT (B-frag over d)
  const __bf16* pk = kp + (size_t)lm * HD + lh * 8;        // += t0*HD per block
  const __bf16* pv = vp + (size_t)lm * SS + lh * 8;        // += 32 per block

  v8f accO[4] = {};
  const float scale = 0.125f;           // 1/sqrt(64)
  int nTb = (s0 + 16 + 31) >> 5;        // causal: t <= s0+15
#pragma unroll 1
  for (int tb = 0; tb < nTb; tb++) {
    int t0 = tb * 32;
    // scores for two 16-wide t halves, then mask+silu -> LDS bf16 tile
#pragma unroll
    for (int th = 0; th < 2; th++) {
      v8f sc = {};
#pragma unroll
      for (int ks = 0; ks < 2; ks++) {
        const __bf16* p = pk + (size_t)th * 16 * HD + ks * 32;
        v8bf lo = *(const v8bf*)p;
        v8bf hi = *(const v8bf*)(p + 16);
        v16bf bf = SHUF16(lo, hi);
        sc = __builtin_amdgcn_wmma_f32_16x16x32_bf16(
            false, qfrag[ks], false, bf, (short)0, sc, false, false);
      }
#pragma unroll
      for (int e = 0; e < 8; e++) {
        int m = e + lh * 8;                 // local q row
        int t = t0 + th * 16 + lm;          // key position
        float x = sc[e] * scale;
        float val = (t <= s0 + m) ? silu_f(x) : 0.0f;
        at[m * 32 + th * 16 + lm] = (__bf16)val;
      }
    }
    // reload as A fragment (same-wave DS ops are in-order)
    const __bf16* pl = at + lm * 32 + lh * 8;
    v8bf alo = *(const v8bf*)pl;
    v8bf ahi = *(const v8bf*)(pl + 16);
    v16bf afrag = SHUF16(alo, ahi);
    // o[16x64] += a[16x32] @ v[32x64]  (vT is N-major: contiguous in t)
#pragma unroll
    for (int j = 0; j < 4; j++) {
      const __bf16* p = pv + (size_t)j * 16 * SS;
      v8bf lo = *(const v8bf*)p;
      v8bf hi = *(const v8bf*)(p + 16);
      v16bf bf = SHUF16(lo, hi);
      accO[j] = __builtin_amdgcn_wmma_f32_16x16x32_bf16(
          false, afrag, false, bf, (short)0, accO[j], false, false);
    }
    pk += 32 * HD;
    pv += 32;
  }
  // write o in [B,S,H*hd] layout (fp32)
#pragma unroll
  for (int j = 0; j < 4; j++)
#pragma unroll
    for (int e = 0; e < 8; e++) {
      int s = s0 + e + lh * 8;
      int d = j * 16 + lm;
      o[((size_t)(b * SS + s)) * DD + h * HD + d] = accO[j][e];
    }
}

// ---- Kernel 3: layernorm(o) * gate -> bf16 -------------------------------
// One wave per row of 512. grid 1024, block 256.
__global__ void ln_gate_kernel(const float* __restrict__ o,
                               const float* __restrict__ gate,
                               const float* __restrict__ g,
                               const float* __restrict__ bln,
                               __bf16* __restrict__ a2) {
  int wid = threadIdx.x >> 5, lane = threadIdx.x & 31;
  int row = blockIdx.x * 8 + wid;
  const float* po = o + (size_t)row * DD;
  float vals[16];
  float sum = 0.0f;
#pragma unroll
  for (int i = 0; i < 16; i++) { vals[i] = po[lane + i * 32]; sum += vals[i]; }
#pragma unroll
  for (int off = 16; off > 0; off >>= 1) sum += __shfl_xor(sum, off, 32);
  float mean = sum * (1.0f / 512.0f);
  float var = 0.0f;
#pragma unroll
  for (int i = 0; i < 16; i++) { float d = vals[i] - mean; var += d * d; }
#pragma unroll
  for (int off = 16; off > 0; off >>= 1) var += __shfl_xor(var, off, 32);
  float r = rsqrtf(var * (1.0f / 512.0f) + 1e-5f);
#pragma unroll
  for (int i = 0; i < 16; i++) {
    int idx = lane + i * 32;
    float nv = (vals[i] - mean) * r * g[idx] + bln[idx];
    a2[(size_t)row * DD + idx] = (__bf16)(nv * gate[(size_t)row * DD + idx]);
  }
}

// ---- Kernel 4: out = a2 @ W2^T + b2 (fp32 out) ---------------------------
// grid (64, 2), block 256.
__global__ __launch_bounds__(256, 1) void gemm2_kernel(
    const __bf16* __restrict__ a2, const __bf16* __restrict__ w2b,
    const float* __restrict__ b2, float* __restrict__ out) {
  int wid = threadIdx.x >> 5, lane = threadIdx.x & 31;
  int row0 = blockIdx.x * 128 + (wid & 1) * 64;
  int col0 = blockIdx.y * 256 + (wid >> 1) * 64;
  v8f acc[4][4] = {};
  wave_gemm_64x64<DD>(a2, w2b, row0, col0, lane, acc);
  int lm = lane & 15, lh = lane >> 4;
#pragma unroll
  for (int j = 0; j < 4; j++) {
    int n = col0 + j * 16 + lm;
    float bias = b2[n];
#pragma unroll
    for (int i = 0; i < 4; i++)
#pragma unroll
      for (int e = 0; e < 8; e++) {
        int bs = row0 + i * 16 + e + lh * 8;
        out[(size_t)bs * DD + n] = acc[i][j][e] + bias;
      }
  }
}

extern "C" void kernel_launch(void* const* d_in, const int* in_sizes, int n_in,
                              void* d_out, int out_size, void* d_ws, size_t ws_size,
                              hipStream_t stream) {
  const float* x    = (const float*)d_in[0];
  const float* W1   = (const float*)d_in[1];
  const float* b1   = (const float*)d_in[2];
  const float* ln_g = (const float*)d_in[3];
  const float* ln_b = (const float*)d_in[4];
  const float* W2   = (const float*)d_in[5];
  const float* b2   = (const float*)d_in[6];
  float* out = (float*)d_out;

  char* ws = (char*)d_ws;
  size_t off = 0;
  __bf16* xb  = (__bf16*)(ws + off); off += (size_t)BS * DD * 2;        // 8 MB
  __bf16* w1b = (__bf16*)(ws + off); off += (size_t)4 * DD * DD * 2;    // 2 MB
  __bf16* w2b = (__bf16*)(ws + off); off += (size_t)DD * DD * 2;        // 0.5 MB
  __bf16* qb  = (__bf16*)(ws + off); off += (size_t)BS * DD * 2;
  __bf16* kb  = (__bf16*)(ws + off); off += (size_t)BS * DD * 2;
  __bf16* vtb = (__bf16*)(ws + off); off += (size_t)BS * DD * 2;
  float*  gatef = (float*)(ws + off); off += (size_t)BS * DD * 4;       // 16 MB
  float*  of    = (float*)(ws + off); off += (size_t)BS * DD * 4;       // 16 MB
  __bf16* a2b = (__bf16*)(ws + off); off += (size_t)BS * DD * 2;

  // 0) converts
  {
    int n0 = BS * DD;          // x: 4,194,304
    int n1 = 4 * DD * DD;      // W1: 1,048,576
    int n2 = DD * DD;          // W2: 262,144
    cvt_bf16_kernel<<<(n0 + 255) / 256, 256, 0, stream>>>(x, xb, n0);
    cvt_bf16_kernel<<<(n1 + 255) / 256, 256, 0, stream>>>(W1, w1b, n1);
    cvt_bf16_kernel<<<(n2 + 255) / 256, 256, 0, stream>>>(W2, w2b, n2);
  }
  // 1) fused qkv+gate projection with SiLU
  gemm1_kernel<<<dim3(BS / 128, (4 * DD) / 256), 256, 0, stream>>>(
      xb, w1b, b1, qb, kb, vtb, gatef);
  // 2) causal SiLU attention
  attn_kernel<<<(BB * HH * (SS / 16)) / 8, 256, 0, stream>>>(qb, kb, vtb, of);
  // 3) layernorm * gate
  ln_gate_kernel<<<BS / 8, 256, 0, stream>>>(of, gatef, ln_g, ln_b, a2b);
  // 4) output projection
  gemm2_kernel<<<dim3(BS / 128, DD / 256), 256, 0, stream>>>(a2b, w2b, b2, out);
}
